// MOLCT_PLUS_24043226923196
// MI455X (gfx1250) — matrix operand order repaired and checked by reading, not compiled, not run on hardware
//
#include <hip/hip_runtime.h>
#include <hip/hip_bf16.h>

typedef __attribute__((ext_vector_type(16))) _Float16 v16h;
typedef __attribute__((ext_vector_type(8)))  float    v8f;

namespace {
constexpr int B_ = 16, A_ = 128, C_ = 64, CP_ = 16, DA_ = 128, DP_ = 64, H_ = 8, DH_ = 16;
constexpr float EPS_ = 1e-5f;
}

__device__ __forceinline__ float wave_sum(float v) {
  for (int off = 16; off > 0; off >>= 1) v += __shfl_xor(v, off, 32);
  return v;
}
__device__ __forceinline__ float wave_max(float v) {
  for (int off = 16; off > 0; off >>= 1) v = fmaxf(v, __shfl_xor(v, off, 32));
  return v;
}

// ---------------------------------------------------------------------------
// Generic batched WMMA GEMM: D = act(alpha * A(MxK) @ B(KxN) + bias), optional
// accumulate. One wave per 16x16 tile; K-loop step 32 via
// v_wmma_f32_16x16x32_f16 (f16 inputs converted from f32, f32 accumulation).
// Compile-time flags:
//   BT    : B is stored transposed (element [k][n] at Bm[n*ldb + k])
//   KG    : K not a multiple of 32 -> clamped-index loads + select-to-zero
//   NG    : N not a multiple of 16 -> clamp column, guard store
//   ACC   : accumulate into C
//   RELU  : apply relu in epilogue
//   BIAS  : add per-column bias in epilogue
// M is always a multiple of 16 for every call in this model (no M guard).
// Batch offset = (bz/bdiv)*s?1 + (bz%bdiv)*s?2  (lets us batch over (b,h)).
// ---------------------------------------------------------------------------
template <int BT, int KG, int NG, int ACC, int RELU, int BIAS>
__global__ __launch_bounds__(32)
void gemm_wmma(const float* __restrict__ A, const float* __restrict__ Bm,
               const float* __restrict__ bias, float* __restrict__ C,
               int N, int K, int lda, int ldb, int ldc,
               long long sA1, long long sA2, long long sB1, long long sB2,
               long long sC1, long long sC2, int bdiv, float alpha) {
  const int lane = threadIdx.x & 31;
  const int n0 = blockIdx.x * 16;
  const int m0 = blockIdx.y * 16;
  const int bz = blockIdx.z;
  const int b1 = bz / bdiv, b2 = bz % bdiv;
  A  += b1 * sA1 + b2 * sA2;
  Bm += b1 * sB1 + b2 * sB2;
  C  += b1 * sC1 + b2 * sC2;
  const int l15 = lane & 15;
  const int kh  = lane >> 4;
  const int mrow  = m0 + l15;
  const int ncolr = n0 + l15;
  const int ncol  = NG ? min(ncolr, N - 1) : ncolr;  // clamped: loads always valid

  v8f acc = {};
  for (int k0 = 0; k0 < K; k0 += 32) {
    v16h af, bf;
    if (KG) {
      // branch-free guarded path: clamp index, select 0 for OOB K
#pragma unroll
      for (int e = 0; e < 16; ++e) {
        int kkA = k0 + (kh << 4) + e;
        float xa = A[(long long)mrow * lda + min(kkA, K - 1)];
        af[e] = (_Float16)(kkA < K ? xa : 0.f);
        int kkB = k0 + (kh << 4) + e;
        int kB = min(kkB, K - 1);
        float xb = BT ? Bm[(long long)ncol * ldb + kB]
                      : Bm[(long long)kB * ldb + ncol];
        bf[e] = (_Float16)(kkB < K ? xb : 0.f);
      }
    } else {
      // fast path: fully unguarded vector loads
      // A layout: af[e] <-> K = k0 + kh*16 + e
      const float* ap = A + (long long)mrow * lda + k0 + (kh << 4);
      float4 a0 = *(const float4*)(ap + 0);
      float4 a1 = *(const float4*)(ap + 4);
      float4 a2 = *(const float4*)(ap + 8);
      float4 a3 = *(const float4*)(ap + 12);
      af[0] = (_Float16)a0.x;  af[1] = (_Float16)a0.y;
      af[2] = (_Float16)a0.z;  af[3] = (_Float16)a0.w;
      af[4] = (_Float16)a1.x;  af[5] = (_Float16)a1.y;
      af[6] = (_Float16)a1.z;  af[7] = (_Float16)a1.w;
      af[8]  = (_Float16)a2.x; af[9]  = (_Float16)a2.y;
      af[10] = (_Float16)a2.z; af[11] = (_Float16)a2.w;
      af[12] = (_Float16)a3.x; af[13] = (_Float16)a3.y;
      af[14] = (_Float16)a3.z; af[15] = (_Float16)a3.w;
      // B layout: bf[e] <-> K = k0 + kh*16 + e
      if (BT) {
        const float* bp = Bm + (long long)ncol * ldb + k0 + (kh << 4);
        float4 b0 = *(const float4*)(bp + 0);
        float4 b1 = *(const float4*)(bp + 4);
        float4 b2 = *(const float4*)(bp + 8);
        float4 b3 = *(const float4*)(bp + 12);
        bf[0] = (_Float16)b0.x;  bf[1] = (_Float16)b0.y;
        bf[2] = (_Float16)b0.z;  bf[3] = (_Float16)b0.w;
        bf[4] = (_Float16)b1.x;  bf[5] = (_Float16)b1.y;
        bf[6] = (_Float16)b1.z;  bf[7] = (_Float16)b1.w;
        bf[8]  = (_Float16)b2.x; bf[9]  = (_Float16)b2.y;
        bf[10] = (_Float16)b2.z; bf[11] = (_Float16)b2.w;
        bf[12] = (_Float16)b3.x; bf[13] = (_Float16)b3.y;
        bf[14] = (_Float16)b3.z; bf[15] = (_Float16)b3.w;
      } else {
        const float* bp = Bm + (long long)(k0 + (kh << 4)) * ldb + ncol;
#pragma unroll
        for (int e = 0; e < 16; ++e) bf[e] = (_Float16)bp[(long long)e * ldb];
      }
    }
    acc = __builtin_amdgcn_wmma_f32_16x16x32_f16(false, af, false, bf,
                                                 (short)0, acc, false, false);
  }
#pragma unroll
  for (int r = 0; r < 8; ++r) {
    int mm = m0 + r + (kh << 3);  // C/D: M = r + 8*(lane>>4), N = lane&15
    float v = acc[r] * alpha;
    if (BIAS) v += bias[ncol];
    if (RELU) v = fmaxf(v, 0.f);
    if (!NG || ncolr < N) {
      long long idx = (long long)mm * ldc + ncolr;
      if (ACC) C[idx] += v; else C[idx] = v;
    }
  }
}

// --------------------------- elementwise kernels ---------------------------
__global__ void k_zero(float* __restrict__ p, long long n) {
  long long i = (long long)blockIdx.x * blockDim.x + threadIdx.x;
  long long st = (long long)gridDim.x * blockDim.x;
  for (; i < n; i += st) p[i] = 0.f;
}

// P[b,i,j,c] += L[b,i,c] + R[b,j,c]
__global__ void k_add_lr(float* __restrict__ P, const float* __restrict__ L,
                         const float* __restrict__ R) {
  const long long n = (long long)B_ * A_ * A_ * DP_;
  long long i = (long long)blockIdx.x * blockDim.x + threadIdx.x;
  long long st = (long long)gridDim.x * blockDim.x;
  for (; i < n; i += st) {
    int c = (int)(i & (DP_ - 1));
    long long row = i >> 6;              // (b*A + ii)*A + j
    int j = (int)(row & (A_ - 1));
    long long bi = row >> 7;             // b*A + ii
    int b = (int)(bi >> 7);
    P[i] += L[bi * DP_ + c] + R[((long long)(b * A_ + j)) * DP_ + c];
  }
}

// O[b,i,j,c] = L[b,i,c] * R[b,j,c]
__global__ void k_outer(float* __restrict__ O, const float* __restrict__ L,
                        const float* __restrict__ R) {
  const long long n = (long long)B_ * A_ * A_ * DP_;
  long long i = (long long)blockIdx.x * blockDim.x + threadIdx.x;
  long long st = (long long)gridDim.x * blockDim.x;
  for (; i < n; i += st) {
    int c = (int)(i & (DP_ - 1));
    long long row = i >> 6;
    int j = (int)(row & (A_ - 1));
    long long bi = row >> 7;
    int b = (int)(bi >> 7);
    O[i] = L[bi * DP_ + c] * R[((long long)(b * A_ + j)) * DP_ + c];
  }
}

// logits[b,h,i,j] += bterm[b,i,j,h]; mask with -1e9
__global__ void k_logits_post(float* __restrict__ Lg, const float* __restrict__ bt,
                              const float* __restrict__ mij) {
  const long long n = (long long)B_ * H_ * A_ * A_;
  long long i = (long long)blockIdx.x * blockDim.x + threadIdx.x;
  long long st = (long long)gridDim.x * blockDim.x;
  for (; i < n; i += st) {
    int j = (int)(i & 127);
    int ii = (int)((i >> 7) & 127);
    int h = (int)((i >> 14) & 7);
    int b = (int)(i >> 17);
    long long pidx = (((long long)b * A_ + ii) * A_ + j);
    float v = Lg[i] + bt[pidx * H_ + h];
    Lg[i] = (mij[pidx] > 0.f) ? v : -1e9f;
  }
}

// softmax over rows of 128 (one wave per row, 4 elems/lane)
__global__ void k_softmax128(float* __restrict__ Lg, int rows) {
  int wave = threadIdx.x >> 5, lane = threadIdx.x & 31;
  int row = blockIdx.x * (blockDim.x >> 5) + wave;
  if (row >= rows) return;
  float* p = Lg + (long long)row * 128;
  float v[4];
  float mx = -1e30f;
#pragma unroll
  for (int t = 0; t < 4; ++t) { v[t] = p[lane + 32 * t]; mx = fmaxf(mx, v[t]); }
  mx = wave_max(mx);
  float s = 0.f;
#pragma unroll
  for (int t = 0; t < 4; ++t) { v[t] = __expf(v[t] - mx); s += v[t]; }
  s = wave_sum(s);
  float inv = 1.f / s;
#pragma unroll
  for (int t = 0; t < 4; ++t) p[lane + 32 * t] = v[t] * inv;
}

// sm[row,c] = s[row,c] * m[row]   (cols = 128)
__global__ void k_mask_rows(float* __restrict__ O, const float* __restrict__ S,
                            const float* __restrict__ m, long long n) {
  long long i = (long long)blockIdx.x * blockDim.x + threadIdx.x;
  long long st = (long long)gridDim.x * blockDim.x;
  for (; i < n; i += st) O[i] = S[i] * m[i >> 7];
}

// acc = (acc + d)*m ; act = LN(act + d)*m   (atom rows: 128 cols, 4/lane)
__global__ void k_fuse_atom(float* __restrict__ act, const float* __restrict__ d,
                            float* __restrict__ acc, const float* __restrict__ mrow,
                            int apply_mask) {
  int wave = threadIdx.x >> 5, lane = threadIdx.x & 31;
  int row = blockIdx.x * (blockDim.x >> 5) + wave;
  if (row >= B_ * A_) return;
  long long base = (long long)row * DA_;
  float m = apply_mask ? mrow[row] : 1.f;
  float x[4], s = 0.f, s2 = 0.f;
#pragma unroll
  for (int t = 0; t < 4; ++t) {
    int c = lane + 32 * t;
    float dd = d[base + c];
    float a = acc[base + c] + dd;
    acc[base + c] = a * m;
    float y = act[base + c] + dd;
    x[t] = y; s += y; s2 += y * y;
  }
  s = wave_sum(s); s2 = wave_sum(s2);
  float mu = s / DA_;
  float var = s2 / DA_ - mu * mu;
  float inv = rsqrtf(var + EPS_);
#pragma unroll
  for (int t = 0; t < 4; ++t) act[base + lane + 32 * t] = (x[t] - mu) * inv * m;
}

// same for pair rows: 64 cols, 2/lane
__global__ void k_fuse_pair(float* __restrict__ act, const float* __restrict__ d,
                            float* __restrict__ acc, const float* __restrict__ mij,
                            int apply_mask) {
  int wave = threadIdx.x >> 5, lane = threadIdx.x & 31;
  long long row = (long long)blockIdx.x * 4 + wave;
  if (row >= (long long)B_ * A_ * A_) return;
  long long base = row * DP_;
  float m = apply_mask ? mij[row] : 1.f;
  float x[2], s = 0.f, s2 = 0.f;
#pragma unroll
  for (int t = 0; t < 2; ++t) {
    int c = lane + 32 * t;
    float dd = d[base + c];
    float a = acc[base + c] + dd;
    acc[base + c] = a * m;
    float y = act[base + c] + dd;
    x[t] = y; s += y; s2 += y * y;
  }
  s = wave_sum(s); s2 = wave_sum(s2);
  float mu = s / DP_;
  float var = s2 / DP_ - mu * mu;
  float inv = rsqrtf(var + EPS_);
#pragma unroll
  for (int t = 0; t < 2; ++t) act[base + lane + 32 * t] = (x[t] - mu) * inv * m;
}

// out += LN(acc)  (atom rows of 128)
__global__ void k_final_atom(float* __restrict__ out, const float* __restrict__ acc) {
  int wave = threadIdx.x >> 5, lane = threadIdx.x & 31;
  int row = blockIdx.x * (blockDim.x >> 5) + wave;
  if (row >= B_ * A_) return;
  long long base = (long long)row * DA_;
  float x[4], s = 0.f, s2 = 0.f;
#pragma unroll
  for (int t = 0; t < 4; ++t) {
    float y = acc[base + lane + 32 * t];
    x[t] = y; s += y; s2 += y * y;
  }
  s = wave_sum(s); s2 = wave_sum(s2);
  float mu = s / DA_;
  float inv = rsqrtf(s2 / DA_ - mu * mu + EPS_);
#pragma unroll
  for (int t = 0; t < 4; ++t) out[base + lane + 32 * t] += (x[t] - mu) * inv;
}

// out += LN(acc)  (pair rows of 64)
__global__ void k_final_pair(float* __restrict__ out, const float* __restrict__ acc) {
  int wave = threadIdx.x >> 5, lane = threadIdx.x & 31;
  long long row = (long long)blockIdx.x * 4 + wave;
  if (row >= (long long)B_ * A_ * A_) return;
  long long base = row * DP_;
  float x[2], s = 0.f, s2 = 0.f;
#pragma unroll
  for (int t = 0; t < 2; ++t) {
    float y = acc[base + lane + 32 * t];
    x[t] = y; s += y; s2 += y * y;
  }
  s = wave_sum(s); s2 = wave_sum(s2);
  float mu = s / DP_;
  float inv = rsqrtf(s2 / DP_ - mu * mu + EPS_);
#pragma unroll
  for (int t = 0; t < 2; ++t) out[base + lane + 32 * t] += (x[t] - mu) * inv;
}

// ------------------------------- host side ---------------------------------
template <int BT, int KG, int NG, int ACC, int RELU, int BIAS>
static inline void gemmT(hipStream_t st, const float* A, const float* Bm, const float* bias,
                         float* Cc, int M, int N, int K, int lda, int ldb, int ldc,
                         long long sA1, long long sA2, long long sB1, long long sB2,
                         long long sC1, long long sC2, int batch, int bdiv, float alpha) {
  dim3 g((N + 15) / 16, (M + 15) / 16, batch);
  gemm_wmma<BT, KG, NG, ACC, RELU, BIAS><<<g, 32, 0, st>>>(
      A, Bm, bias, Cc, N, K, lda, ldb, ldc,
      sA1, sA2, sB1, sB2, sC1, sC2, bdiv, alpha);
}

extern "C" void kernel_launch(void* const* d_in, const int* in_sizes, int n_in,
                              void* d_out, int out_size, void* d_ws, size_t ws_size,
                              hipStream_t stream) {
  (void)in_sizes; (void)n_in; (void)out_size;
  // params flattened in dict-insertion order: 4x8 FT tensors, then 3x12 layer tensors
  const float* P[68];
  for (int i = 0; i < 68; ++i) P[i] = (const float*)d_in[i];
  const float* ar  = (const float*)d_in[68];  // [B,A,C]
  const float* mi  = (const float*)d_in[69];  // [B,A]
  const float* pr  = (const float*)d_in[70];  // [B,A,A,CP]
  const float* mij = (const float*)d_in[71];  // [B,A,A]

  auto FT = [&](int l, int j) { return P[8 * l + j]; };       // Wa,ba,Wp,bp,Wl,bl,Wr,br
  auto LY = [&](int l, int j) { return P[32 + 12 * l + j]; }; // Wq,Wk,Wv,Wo,Wb,Wt1,Wt2,Wopl,Wopr,Wopo,puWt1,puWt2

  const long long NEED = 73924608LL;
  if (ws_size < (size_t)NEED * sizeof(float)) return;
  float* w = (float*)d_ws;
  auto take = [&](long long n) { float* p = w; w += n; return p; };
  float* region1 = take(33554432); // lr [262144,64] then tpair [262144,128]
  float* region2 = take(16777216); // dpair [262144,64]
  float* accP    = take(16777216);
  float* accA    = take(262144);
  float* q   = take(262144);
  float* kk  = take(262144);
  float* vv  = take(262144);
  float* oo  = take(262144);
  float* smb = take(262144);
  float* lft = take(131072);
  float* rgt = take(131072);
  float* bterm = take(2097152);   // [B,A,A,H]
  float* logit = take(2097152);   // [B,H,A,A]
  float* tatom = take(524288);    // [2048,256]
  float* datom = take(262144);    // [2048,128]

  float* outA = (float*)d_out;           // atom_feat [B,A,DA]
  float* outP = outA + 262144;           // pair_feat [B,A,A,DP]

  const long long nPair = (long long)B_ * A_ * A_ * DP_;

  // ---- FeatureTransformer 0 -> accumulators ----
  gemmT<0,0,0,0,0,1>(stream, ar, FT(0,0), FT(0,1), accA, 2048,128,64, 64,128,128, 0,0,0,0,0,0, 1,1, 1.f);
  gemmT<0,1,0,0,0,1>(stream, pr, FT(0,2), FT(0,3), accP, 262144,64,16, 16,64,64, 0,0,0,0,0,0, 1,1, 1.f);
  gemmT<0,0,0,0,0,1>(stream, ar, FT(0,4), FT(0,5), lft, 2048,64,64, 64,64,64, 0,0,0,0,0,0, 1,1, 1.f);
  gemmT<0,0,0,0,0,1>(stream, ar, FT(0,6), FT(0,7), rgt, 2048,64,64, 64,64,64, 0,0,0,0,0,0, 1,1, 1.f);
  k_add_lr<<<4096,256,0,stream>>>(accP, lft, rgt);
  k_zero<<<256,256,0,stream>>>(outA, 262144);
  k_zero<<<4096,256,0,stream>>>(outP, nPair);

  for (int l = 0; l < 3; ++l) {
    // ---- FeatureTransformer l+1 accumulated into feats ----
    gemmT<0,0,0,1,0,1>(stream, ar, FT(l+1,0), FT(l+1,1), outA, 2048,128,64, 64,128,128, 0,0,0,0,0,0, 1,1, 1.f);
    gemmT<0,1,0,1,0,1>(stream, pr, FT(l+1,2), FT(l+1,3), outP, 262144,64,16, 16,64,64, 0,0,0,0,0,0, 1,1, 1.f);
    gemmT<0,0,0,0,0,1>(stream, ar, FT(l+1,4), FT(l+1,5), lft, 2048,64,64, 64,64,64, 0,0,0,0,0,0, 1,1, 1.f);
    gemmT<0,0,0,0,0,1>(stream, ar, FT(l+1,6), FT(l+1,7), rgt, 2048,64,64, 64,64,64, 0,0,0,0,0,0, 1,1, 1.f);
    k_add_lr<<<4096,256,0,stream>>>(outP, lft, rgt);

    // ---- atom update: pair-biased MHA ----
    gemmT<0,0,0,0,0,0>(stream, outA, LY(l,0), nullptr, q,  2048,128,128, 128,128,128, 0,0,0,0,0,0, 1,1, 1.f);
    gemmT<0,0,0,0,0,0>(stream, outA, LY(l,1), nullptr, kk, 2048,128,128, 128,128,128, 0,0,0,0,0,0, 1,1, 1.f);
    gemmT<0,0,0,0,0,0>(stream, outA, LY(l,2), nullptr, vv, 2048,128,128, 128,128,128, 0,0,0,0,0,0, 1,1, 1.f);
    gemmT<0,0,1,0,0,0>(stream, outP, LY(l,4), nullptr, bterm, 262144,8,64, 64,8,8, 0,0,0,0,0,0, 1,1, 1.f);
    // logits[b,h] = 0.25 * q(b,:,h,:) @ k(b,:,h,:)^T   (batch over (b,h))
    gemmT<1,1,0,0,0,0>(stream, q, kk, nullptr, logit, 128,128,16, 128,128,128,
                       (long long)A_*DA_, DH_, (long long)A_*DA_, DH_,
                       (long long)H_*A_*A_, (long long)A_*A_, B_*H_, H_, 0.25f);
    k_logits_post<<<2048,256,0,stream>>>(logit, bterm, mij);
    k_softmax128<<<(B_*H_*A_)/4,128,0,stream>>>(logit, B_*H_*A_);
    // o[b,:,h,:] = attn[b,h] @ v(b,:,h,:)
    gemmT<0,0,0,0,0,0>(stream, logit, vv, nullptr, oo, 128,16,128, 128,128,128,
                       (long long)H_*A_*A_, (long long)A_*A_, (long long)A_*DA_, DH_,
                       (long long)A_*DA_, DH_, B_*H_, H_, 1.f);
    gemmT<0,0,0,0,0,0>(stream, oo, LY(l,3), nullptr, datom, 2048,128,128, 128,128,128, 0,0,0,0,0,0, 1,1, 1.f);
    k_fuse_atom<<<512,128,0,stream>>>(outA, datom, accA, mi, 0);
    // atom transition
    gemmT<0,0,0,0,1,0>(stream, outA, LY(l,5), nullptr, tatom, 2048,256,128, 128,256,256, 0,0,0,0,0,0, 1,1, 1.f);
    gemmT<0,0,0,0,0,0>(stream, tatom, LY(l,6), nullptr, datom, 2048,128,256, 256,128,128, 0,0,0,0,0,0, 1,1, 1.f);
    k_fuse_atom<<<512,128,0,stream>>>(outA, datom, accA, mi, 1);

    // ---- pair update: outer product + transition ----
    k_mask_rows<<<256,256,0,stream>>>(smb, outA, mi, 262144);
    gemmT<0,0,0,0,0,0>(stream, smb, LY(l,7), nullptr, lft, 2048,64,128, 128,64,64, 0,0,0,0,0,0, 1,1, 1.f);
    gemmT<0,0,0,0,0,0>(stream, smb, LY(l,8), nullptr, rgt, 2048,64,128, 128,64,64, 0,0,0,0,0,0, 1,1, 1.f);
    k_outer<<<4096,256,0,stream>>>(region1, lft, rgt);
    gemmT<0,0,0,0,0,0>(stream, region1, LY(l,9), nullptr, region2, 262144,64,64, 64,64,64, 0,0,0,0,0,0, 1,1, 1.f);
    k_fuse_pair<<<65536,128,0,stream>>>(outP, region2, accP, mij, 0);
    // pair transition (region1 reused as [262144,128] intermediate)
    gemmT<0,0,0,0,1,0>(stream, outP, LY(l,10), nullptr, region1, 262144,128,64, 64,128,128, 0,0,0,0,0,0, 1,1, 1.f);
    gemmT<0,0,0,0,0,0>(stream, region1, LY(l,11), nullptr, region2, 262144,64,128, 128,64,64, 0,0,0,0,0,0, 1,1, 1.f);
    k_fuse_pair<<<65536,128,0,stream>>>(outP, region2, accP, mij, 1);
  }

  // ---- final ResiDual merge ----
  k_final_atom<<<512,128,0,stream>>>(outA, accA);
  k_final_pair<<<65536,128,0,stream>>>(outP, accP);
}
